// GAT_60120952209845
// MI455X (gfx1250) — matrix operand order repaired and checked by reading, not compiled, not run on hardware
//
#include <hip/hip_runtime.h>
#include <hip/hip_bf16.h>

#define NNODES 40000
#define NEDGES 640000
#define DH 128

typedef __attribute__((ext_vector_type(16))) __bf16 v16bf;
typedef __attribute__((ext_vector_type(8)))  float  v8f;

// ---------- helpers ----------
__device__ __forceinline__ unsigned short f2bf(float f) {
    unsigned u = __float_as_uint(f);
    unsigned r = u + 0x7FFFu + ((u >> 16) & 1u);   // round-to-nearest-even
    return (unsigned short)(r >> 16);
}
// monotone int encoding of float for atomicMax-based segment max
__device__ __forceinline__ int fenc(float f) {
    int i = __float_as_int(f);
    return i >= 0 ? i : (i ^ 0x7fffffff);
}
__device__ __forceinline__ float fdec(int i) {
    return __int_as_float(i >= 0 ? i : (i ^ 0x7fffffff));
}

// ---------- pack W (128x128 f32, row-major) -> bf16 column-major ----------
__global__ void gat_packW(const float* __restrict__ W, unsigned short* __restrict__ wt) {
    int i = blockIdx.x * blockDim.x + threadIdx.x;
    if (i >= DH * DH) return;
    int k = i >> 7, col = i & 127;
    wt[col * DH + k] = f2bf(W[k * DH + col]);
}

// ---------- stream-convert activations fp32 -> bf16 (fused inter-layer ReLU) ----------
__global__ __launch_bounds__(256)
void gat_tobf16(const float* __restrict__ in, unsigned short* __restrict__ out,
                int n4, int relu) {
    int i = blockIdx.x * blockDim.x + threadIdx.x;
    if (i >= n4) return;
    float4 v = ((const float4*)in)[i];
    if (relu) {
        v.x = fmaxf(v.x, 0.f); v.y = fmaxf(v.y, 0.f);
        v.z = fmaxf(v.z, 0.f); v.w = fmaxf(v.w, 0.f);
    }
    union { unsigned short s[4]; uint2 q; } u;
    u.s[0] = f2bf(v.x); u.s[1] = f2bf(v.y); u.s[2] = f2bf(v.z); u.s[3] = f2bf(v.w);
    ((uint2*)out)[i] = u.q;
}

// ---------- init aggregation buffer with bias, reset max/denom ----------
__global__ void gat_init(float* __restrict__ out, const float* __restrict__ b,
                         int* __restrict__ menc, float* __restrict__ denom, int n) {
    int i = blockIdx.x * blockDim.x + threadIdx.x;
    if (i < n * DH) out[i] = b[i & 127];
    if (i < n) { menc[i] = (int)0x80000000; denom[i] = 0.f; }
}

// ---------- WMMA GEMM: h = abf @ W  (bf16 inputs, f32 accum) ----------
// block = 256 threads = 8 waves; wave owns a 16x128 strip; per-wave:
// preload 4 A fragments (K=0..127), then 8 column tiles x 4 WMMA each.
__global__ __launch_bounds__(256)
void gat_gemm_wmma(const unsigned short* __restrict__ abf,
                   const unsigned short* __restrict__ wt,
                   float* __restrict__ h, int n) {
    const int lane = threadIdx.x & 31;
    const int wave = threadIdx.x >> 5;
    const int half = lane >> 4;        // 0: lanes 0-15, 1: lanes 16-31
    const int l16  = lane & 15;
    const int rowBase = blockIdx.x * 128 + wave * 16;

    // wave-uniform early exit (EXEC stays all-ones in surviving waves)
    if (rowBase >= n) return;
    const bool fullWave = (rowBase + 16 <= n);

    // A row (clamped: OOB rows feed only never-stored outputs)
    const int arow = min(rowBase + l16, n - 1);
    const unsigned short* ap = abf + (size_t)arow * DH;

    // ---- preload 4 A fragments (16x32 bf16 each, ISA lane layout) ----
    union AF { v16bf v; uint4 q[2]; };
    AF a[4];
#pragma unroll
    for (int kk = 0; kk < 4; ++kk) {
        const int ka = kk * 32 + half * 8;                 // multiple of 8 -> 16B aligned
        a[kk].q[0] = *(const uint4*)(ap + ka);             // K = ka..ka+7
        a[kk].q[1] = *(const uint4*)(ap + ka + 16);        // K = ka+16..ka+23
    }

    const int rout0 = rowBase + half * 8;
#pragma unroll
    for (int ct = 0; ct < 8; ++ct) {
        const int col = ct * 16 + l16;
        const unsigned short* bp = wt + col * DH + half * 8;
        v8f acc = {};
#pragma unroll
        for (int kk = 0; kk < 4; ++kk) {
            union { v16bf v; uint4 q[2]; } b;
            b.q[0] = *(const uint4*)(bp + kk * 32);        // K = k0+8h..k0+8h+7
            b.q[1] = *(const uint4*)(bp + kk * 32 + 16);   // K = k0+16+8h..k0+16+8h+7
            acc = __builtin_amdgcn_wmma_f32_16x16x32_bf16(
                false, a[kk].v, false, b.v, (short)0, acc, false, false);
        }
        // C layout: VGPR i -> M = rout0 + i, N = col
        // one base pointer per lane; 8 stores at compile-time offsets i*DH*4
        float* hp = h + (size_t)rout0 * DH + col;
        if (fullWave) {
#pragma unroll
            for (int i = 0; i < 8; ++i) hp[i * DH] = acc[i];
        } else {
#pragma unroll
            for (int i = 0; i < 8; ++i)
                if (rout0 + i < n) hp[i * DH] = acc[i];
        }
    }
}

// ---------- per-node attention logits: e_src = h.a_s, e_dst = h.a_d ----------
__global__ __launch_bounds__(256)
void gat_logits(const float* __restrict__ h, const float* __restrict__ a_s,
                const float* __restrict__ a_d, float* __restrict__ es,
                float* __restrict__ ed, int n) {
    int node = (int)((blockIdx.x * blockDim.x + threadIdx.x) >> 5);
    int lane = threadIdx.x & 31;
    if (node >= n) return;
    const float4 hv = ((const float4*)(h + (size_t)node * DH))[lane];
    const float4 sv = ((const float4*)a_s)[lane];
    const float4 dv = ((const float4*)a_d)[lane];
    float ps = hv.x * sv.x + hv.y * sv.y + hv.z * sv.z + hv.w * sv.w;
    float pd = hv.x * dv.x + hv.y * dv.y + hv.z * dv.z + hv.w * dv.w;
#pragma unroll
    for (int o = 16; o > 0; o >>= 1) {
        ps += __shfl_xor(ps, o, 32);
        pd += __shfl_xor(pd, o, 32);
    }
    if (lane == 0) { es[node] = ps; ed[node] = pd; }
}

// ---------- edge logits + leaky relu + segment max (int-encoded atomicMax) ----------
__global__ void gat_edge_logit_max(const long long* __restrict__ src,
                                   const long long* __restrict__ dst,
                                   const float* __restrict__ es, const float* __restrict__ ed,
                                   float* __restrict__ ee, int* __restrict__ menc, int ne) {
    int e = blockIdx.x * blockDim.x + threadIdx.x;
    if (e >= ne) return;
    int s = (int)src[e], d = (int)dst[e];
    float v = es[s] + ed[d];
    v = v > 0.f ? v : 0.2f * v;           // LeakyReLU(0.2)
    ee[e] = v;
    atomicMax(menc + d, fenc(v));
}

// ---------- ex = exp(e - m[dst]); denom[dst] += ex ----------
__global__ void gat_edge_exp(const long long* __restrict__ dst, float* __restrict__ ee,
                             const int* __restrict__ menc, float* __restrict__ denom, int ne) {
    int e = blockIdx.x * blockDim.x + threadIdx.x;
    if (e >= ne) return;
    int d = (int)dst[e];
    float ex = __expf(ee[e] - fdec(menc[d]));
    ee[e] = ex;
    atomicAdd(denom + d, ex);
}

// ---------- out[dst] += (ex/denom[dst]) * h[src], one wave per edge ----------
__global__ __launch_bounds__(256)
void gat_scatter(const long long* __restrict__ src, const long long* __restrict__ dst,
                 const float* __restrict__ ee, const float* __restrict__ denom,
                 const float* __restrict__ h, float* __restrict__ out, int ne) {
    int e = (int)((blockIdx.x * blockDim.x + threadIdx.x) >> 5);
    int lane = threadIdx.x & 31;
    if (e >= ne) return;
    int s = (int)src[e], d = (int)dst[e];
    float alpha = ee[e] / denom[d];
    const float4 hv = ((const float4*)(h + (size_t)s * DH))[lane];
    float* op = out + (size_t)d * DH + lane * 4;
    atomicAdd(op + 0, alpha * hv.x);
    atomicAdd(op + 1, alpha * hv.y);
    atomicAdd(op + 2, alpha * hv.z);
    atomicAdd(op + 3, alpha * hv.w);
}

extern "C" void kernel_launch(void* const* d_in, const int* in_sizes, int n_in,
                              void* d_out, int out_size, void* d_ws, size_t ws_size,
                              hipStream_t stream) {
    (void)in_sizes; (void)n_in; (void)out_size; (void)ws_size;
    const int N = NNODES, E = NEDGES;

    const float*     x   = (const float*)d_in[0];
    const long long* ei  = (const long long*)d_in[1];   // int64 [2,E]
    const long long* src = ei;
    const long long* dst = ei + E;

    const float* Wp[3] = { (const float*)d_in[2], (const float*)d_in[6],  (const float*)d_in[10] };
    const float* As[3] = { (const float*)d_in[3], (const float*)d_in[7],  (const float*)d_in[11] };
    const float* Ad[3] = { (const float*)d_in[4], (const float*)d_in[8],  (const float*)d_in[12] };
    const float* Bs[3] = { (const float*)d_in[5], (const float*)d_in[9],  (const float*)d_in[13] };

    // workspace layout
    char* ws = (char*)d_ws;
    float*          hbuf  = (float*)(ws);                                     // N*128 f32
    float*          agg   = (float*)(ws + (size_t)N * DH * 4);                // N*128 f32
    unsigned short* abf   = (unsigned short*)(ws + (size_t)2 * N * DH * 4);   // N*128 bf16
    float*          esrc  = (float*)(ws + (size_t)2 * N * DH * 4 + (size_t)N * DH * 2); // N
    float*          edst  = esrc + N;                                         // N
    int*            menc  = (int*)(edst + N);                                 // N
    float*          denom = (float*)(menc + N);                               // N
    float*          eedge = denom + N;                                        // E
    unsigned short* wt    = (unsigned short*)(eedge + E);                     // 128*128 bf16

    // layer in/out chaining: x -> agg -> agg -> d_out
    // (bias-init of `out` runs AFTER tobf16 consumed `in`, so in/out may alias)
    const float* ins[3]  = { x, agg, agg };
    float*       outs[3] = { agg, agg, (float*)d_out };

    for (int L = 0; L < 3; ++L) {
        gat_packW<<<(DH * DH + 255) / 256, 256, 0, stream>>>(Wp[L], wt);
        gat_tobf16<<<(N * (DH / 4) + 255) / 256, 256, 0, stream>>>(ins[L], abf, N * (DH / 4), L > 0);
        gat_gemm_wmma<<<(N + 127) / 128, 256, 0, stream>>>(abf, wt, hbuf, N);
        gat_init<<<((size_t)N * DH + 255) / 256, 256, 0, stream>>>(outs[L], Bs[L], menc, denom, N);
        gat_logits<<<(N + 7) / 8, 256, 0, stream>>>(hbuf, As[L], Ad[L], esrc, edst, N);
        gat_edge_logit_max<<<(E + 255) / 256, 256, 0, stream>>>(src, dst, esrc, edst, eedge, menc, E);
        gat_edge_exp<<<(E + 255) / 256, 256, 0, stream>>>(dst, eedge, menc, denom, E);
        gat_scatter<<<(E + 7) / 8, 256, 0, stream>>>(src, dst, eedge, denom, hbuf, outs[L], E);
    }
}
